// Edge2NodeQKVAttnLayer_8735963480242
// MI455X (gfx1250) — matrix operand order repaired and checked by reading, not compiled, not run on hardware
//
#include <hip/hip_runtime.h>
#include <math.h>

// ---------------------------------------------------------------------------
// Problem constants (B=8, N=128, HN=256, HE=128, H=8, DH=32, P=N(N-1)/2)
// ---------------------------------------------------------------------------
#define CB   8
#define CN   128
#define CHN  256
#define CHE  128
#define CH   8
#define CDH  32
#define CP   8128
#define CBP  (CB * CP)      // 65024
#define CBN  (CB * CN)      // 1024

typedef __attribute__((ext_vector_type(16))) __bf16 v16bf;
typedef __attribute__((ext_vector_type(8)))  float  v8f;

union BF16x16 { v16bf v; unsigned short u[16]; };

#define HAVE_TDM (__has_builtin(__builtin_amdgcn_tensor_load_to_lds) && \
                  __has_builtin(__builtin_amdgcn_s_wait_tensorcnt))

__device__ __forceinline__ float bf2f(unsigned short u) {
    union { unsigned int i; float f; } x; x.i = ((unsigned int)u) << 16; return x.f;
}
__device__ __forceinline__ unsigned short f2bf(float f) {
    union { float f; unsigned int i; } x; x.f = f;
    unsigned int r = x.i + 0x7FFFu + ((x.i >> 16) & 1u);   // round-to-nearest-even
    return (unsigned short)(r >> 16);
}
__device__ __forceinline__ v8f v8f_zero() {
    v8f z;
#pragma unroll
    for (int e = 0; e < 8; ++e) z[e] = 0.0f;
    return z;
}
__device__ __forceinline__ float wave_sum(float s) {
#pragma unroll
    for (int o = 16; o > 0; o >>= 1) s += __shfl_xor(s, o, 32);
    return s;
}
__device__ __forceinline__ float wave_max(float s) {
#pragma unroll
    for (int o = 16; o > 0; o >>= 1) s = fmaxf(s, __shfl_xor(s, o, 32));
    return s;
}

// pair index for i<j in triu ordering with N=128: p = i*(255-i)/2 + (j-i-1)
__device__ __forceinline__ int pair_idx(int i, int j) {
    return (i * (2 * CN - 1 - i)) / 2 + (j - i - 1);
}

// ---------------------------------------------------------------------------
// TDM 2-D bf16 tile load: 32 (K) x 64 (rows) into LDS, with hardware padding
// of 4 DWORDs after every 16 DWORDs so the LDS row stride is 40 halves.
// Descriptor packing per CDNA5 ISA 8.3/8.4 (count=1, type=2, data_size=2B).
// Dims are tile-start-relative; rows beyond tdim1 are zero-filled by TDM.
// ---------------------------------------------------------------------------
#if HAVE_TDM
typedef unsigned int u32x4 __attribute__((ext_vector_type(4)));
typedef int          i32x8 __attribute__((ext_vector_type(8)));
typedef int          i32x4 __attribute__((ext_vector_type(4)));

__device__ __forceinline__ void tdm_load_tile_bf16(
    unsigned lds_addr, const unsigned short* gptr,
    unsigned tdim0, unsigned tdim1, unsigned long long stride_elems)
{
    const unsigned long long ga = (unsigned long long)(uintptr_t)gptr;
    u32x4 g0;
    g0[0] = 1u;                                                // count=1 (valid)
    g0[1] = lds_addr;                                          // LDS byte address
    g0[2] = (unsigned)(ga & 0xFFFFFFFFu);                      // global_addr[31:0]
    g0[3] = (unsigned)((ga >> 32) & 0x1FFFFFFu) | (2u << 30);  // [56:32] | type=2
    i32x8 g1;
    g1[0] = (int)((1u << 16)        // data_size = 2 bytes
                | (1u << 20)        // pad_enable
                | (3u << 22)        // pad_interval: every 16 DWORDs
                | (3u << 25));      // pad_amount: 4 DWORDs
    g1[1] = (int)((tdim0 & 0xFFFFu) << 16);                    // tensor_dim0[15:0]
    g1[2] = (int)(((tdim0 >> 16) & 0xFFFFu) | ((tdim1 & 0xFFFFu) << 16));
    g1[3] = (int)(((tdim1 >> 16) & 0xFFFFu) | (32u << 16));    // tile_dim0 = 32
    g1[4] = 64;                                                // tile_dim1 = 64, tile_dim2 = 0
    g1[5] = (int)(unsigned)(stride_elems & 0xFFFFFFFFull);     // tensor_dim0_stride lo
    g1[6] = (int)(unsigned)((stride_elems >> 32) & 0xFFFFull); // stride hi, dim1_stride = 0
    g1[7] = 0;
    i32x4 gz = {0, 0, 0, 0};
#if defined(__clang_major__) && (__clang_major__ >= 23)
    i32x8 gz8 = {0, 0, 0, 0, 0, 0, 0, 0};
    __builtin_amdgcn_tensor_load_to_lds(g0, g1, gz, gz, gz8, 0);
#else
    __builtin_amdgcn_tensor_load_to_lds(g0, g1, gz, gz, 0);
#endif
}
#endif

// ---------------------------------------------------------------------------
// Weight prep: Wt[n][k] = bf16(W[k][n])  (tiny, runs once, L2-resident after)
// ---------------------------------------------------------------------------
__global__ __launch_bounds__(256) void wt_kernel(
    const float* __restrict__ W, unsigned short* __restrict__ Wt, int K, int Nout)
{
    const int idx = blockIdx.x * 256 + threadIdx.x;
    if (idx >= K * Nout) return;
    const int k = idx / Nout, n = idx % Nout;
    Wt[(size_t)n * K + k] = f2bf(W[idx]);
}

// ---------------------------------------------------------------------------
// LayerNorm of node features -> bf16 into concat buffer nat[1024][512] cols 0..255
// ---------------------------------------------------------------------------
__global__ __launch_bounds__(256) void ln_node_kernel(
    const float* __restrict__ x, const float* __restrict__ g,
    const float* __restrict__ b, unsigned short* __restrict__ nat)
{
    const int lane = threadIdx.x & 31, w = threadIdx.x >> 5;
    const int row = blockIdx.x * 8 + w;
    const float* xr = x + (size_t)row * CHN;
    float v[8]; float s = 0.f;
#pragma unroll
    for (int i = 0; i < 8; ++i) { v[i] = xr[i * 32 + lane]; s += v[i]; }
    const float mean = wave_sum(s) * (1.0f / CHN);
    float q = 0.f;
#pragma unroll
    for (int i = 0; i < 8; ++i) { float d = v[i] - mean; q += d * d; }
    const float inv = rsqrtf(wave_sum(q) * (1.0f / CHN) + 1e-5f);
#pragma unroll
    for (int i = 0; i < 8; ++i) {
        const int c = i * 32 + lane;
        nat[(size_t)row * 512 + c] = f2bf((v[i] - mean) * inv * g[c] + b[c]);
    }
}

// ---------------------------------------------------------------------------
// LayerNorm of edge features -> bf16 e_in[65024][128]
// ---------------------------------------------------------------------------
__global__ __launch_bounds__(256) void ln_edge_kernel(
    const float* __restrict__ x, const float* __restrict__ g,
    const float* __restrict__ b, unsigned short* __restrict__ e_in)
{
    const int lane = threadIdx.x & 31, w = threadIdx.x >> 5;
    const int row = blockIdx.x * 8 + w;
    const float* xr = x + (size_t)row * CHE;
    float v[4]; float s = 0.f;
#pragma unroll
    for (int i = 0; i < 4; ++i) { v[i] = xr[i * 32 + lane]; s += v[i]; }
    const float mean = wave_sum(s) * (1.0f / CHE);
    float q = 0.f;
#pragma unroll
    for (int i = 0; i < 4; ++i) { float d = v[i] - mean; q += d * d; }
    const float inv = rsqrtf(wave_sum(q) * (1.0f / CHE) + 1e-5f);
#pragma unroll
    for (int i = 0; i < 4; ++i) {
        const int c = i * 32 + lane;
        e_in[(size_t)row * CHE + c] = f2bf((v[i] - mean) * inv * g[c] + b[c]);
    }
}

// ---------------------------------------------------------------------------
// Shared WMMA step: 2x2 tiles of 16x16 per wave from staged LDS tiles
// (fragment layouts per ISA 7.12.2)
// ---------------------------------------------------------------------------
#define LDSTRIDE 40
__device__ __forceinline__ void wmma_step(
    const unsigned short (*__restrict__ Asb)[LDSTRIDE],
    const unsigned short (*__restrict__ Bsb)[LDSTRIDE],
    int waveM, int waveN, int mlo, int hi, v8f acc[2][2])
{
    BF16x16 afr[2], bfr[2];
#pragma unroll
    for (int tm = 0; tm < 2; ++tm) {
        const int am = waveM * 32 + tm * 16 + mlo;
        const int kb = hi * 8;
#pragma unroll
        for (int e = 0; e < 16; ++e)
            afr[tm].u[e] = Asb[am][kb + ((e >> 3) << 4) + (e & 7)];
    }
#pragma unroll
    for (int tn = 0; tn < 2; ++tn) {
        const int bn = waveN * 32 + tn * 16 + mlo;
        const int kb = hi * 8;
#pragma unroll
        for (int e = 0; e < 16; ++e)
            bfr[tn].u[e] = Bsb[bn][kb + ((e >> 3) << 4) + (e & 7)];
    }
#pragma unroll
    for (int tm = 0; tm < 2; ++tm)
#pragma unroll
        for (int tn = 0; tn < 2; ++tn)
            acc[tm][tn] = __builtin_amdgcn_wmma_f32_16x16x32_bf16(
                false, afr[tm].v, false, bfr[tn].v,
                (short)0, acc[tm][tn], false, false);
}

// ---------------------------------------------------------------------------
// Generic bf16 WMMA GEMM:  C[M,Nout] = A_bf16[M,K](lda) @ Wt_bf16[Nout,K]^T + bias
// block = 128 threads (4 waves, 2x2), tile 64x64, K-step 32.
// TDM path: wave 0 DMAs double-buffered A/B tiles (tensor_load_to_lds) and the
// next tile's DMA overlaps the current tile's WMMAs.
// mode 0: QKV split (q bf16, k bf16, v f32, [B,H,N,DH]); 1: e_val bf16;
// mode 2: e_adj f32 [M,8]; 3: gelu->comb bf16; 4: sk f32 [M,512]
// ---------------------------------------------------------------------------
__global__ __launch_bounds__(128) void gemm_bf16_kernel(
    const unsigned short* __restrict__ A, int lda,
    const unsigned short* __restrict__ Wt, int Nout,
    const float* __restrict__ bias,
    int M, int K, int mode,
    void* out0, void* out1, void* out2)
{
    __shared__ unsigned short As[2][64][LDSTRIDE];
    __shared__ unsigned short Bs[2][64][LDSTRIDE];

    const int tid  = threadIdx.x;
    const int lane = tid & 31, wid = tid >> 5;
    const int waveM = wid >> 1, waveN = wid & 1;
    const int mlo = lane & 15, hi = lane >> 4;
    const int row0 = blockIdx.x * 64, n0 = blockIdx.y * 64;
    const unsigned bRows = (unsigned)(Nout > n0 ? Nout - n0 : 0);

    v8f acc[2][2];
    acc[0][0] = v8f_zero(); acc[0][1] = v8f_zero();
    acc[1][0] = v8f_zero(); acc[1][1] = v8f_zero();

#if HAVE_TDM
    if (wid == 0) {
        tdm_load_tile_bf16((unsigned)(uintptr_t)&As[0][0][0],
                           A + (size_t)row0 * lda,
                           (unsigned)K, (unsigned)(M - row0), (unsigned long long)lda);
        tdm_load_tile_bf16((unsigned)(uintptr_t)&Bs[0][0][0],
                           Wt + (size_t)n0 * K,
                           (unsigned)K, bRows, (unsigned long long)K);
    }
    int cur = 0;
    for (int k0 = 0; k0 < K; k0 += 32) {
        if (wid == 0) __builtin_amdgcn_s_wait_tensorcnt(0);  // current tiles resident
        __syncthreads();
        if (wid == 0 && (k0 + 32) < K) {                     // DMA next, overlap WMMA
            const int kn = k0 + 32;
            tdm_load_tile_bf16((unsigned)(uintptr_t)&As[cur ^ 1][0][0],
                               A + (size_t)row0 * lda + kn,
                               (unsigned)(K - kn), (unsigned)(M - row0),
                               (unsigned long long)lda);
            tdm_load_tile_bf16((unsigned)(uintptr_t)&Bs[cur ^ 1][0][0],
                               Wt + (size_t)n0 * K + kn,
                               (unsigned)(K - kn), bRows, (unsigned long long)K);
        }
        wmma_step(As[cur], Bs[cur], waveM, waveN, mlo, hi, acc);
        __syncthreads();                                     // tile consumed
        cur ^= 1;
    }
#else
    for (int k0 = 0; k0 < K; k0 += 32) {
        int s = tid * 2;
#pragma unroll
        for (int it = 0; it < 2; ++it, ++s) {
            const int r = s >> 2, sc = (s & 3) * 8;
            const uint4 va = *(const uint4*)(A + (size_t)(row0 + r) * lda + k0 + sc);
            const unsigned short* pa = (const unsigned short*)&va;
#pragma unroll
            for (int e = 0; e < 8; ++e) As[0][r][sc + e] = pa[e];
            if ((unsigned)r < bRows) {
                const uint4 vb = *(const uint4*)(Wt + (size_t)(n0 + r) * K + k0 + sc);
                const unsigned short* pb = (const unsigned short*)&vb;
#pragma unroll
                for (int e = 0; e < 8; ++e) Bs[0][r][sc + e] = pb[e];
            } else {
#pragma unroll
                for (int e = 0; e < 8; ++e) Bs[0][r][sc + e] = 0;
            }
        }
        if (k0 + 32 < K)
            __builtin_prefetch(A + (size_t)(row0 + (tid >> 1)) * lda + k0 + 32, 0, 1);
        __syncthreads();
        wmma_step(As[0], Bs[0], waveM, waveN, mlo, hi, acc);
        __syncthreads();
    }
#endif

    // --- epilogue
#pragma unroll
    for (int tm = 0; tm < 2; ++tm) {
#pragma unroll
        for (int tn = 0; tn < 2; ++tn) {
            const int col = n0 + waveN * 32 + tn * 16 + mlo;
            if (col >= Nout) continue;
            const float bv = bias[col];
#pragma unroll
            for (int r = 0; r < 8; ++r) {
                const int row = row0 + waveM * 32 + tm * 16 + hi * 8 + r;
                if (row >= M) continue;
                const float x = acc[tm][tn][r] + bv;
                if (mode == 0) {
                    const int which = col >> 8, h = (col >> 5) & 7, d = col & 31;
                    const int bb = row >> 7, nn = row & 127;
                    const size_t idx = (((size_t)bb * CH + h) * CN + nn) * CDH + d;
                    if (which == 0)      ((unsigned short*)out0)[idx] = f2bf(x);
                    else if (which == 1) ((unsigned short*)out1)[idx] = f2bf(x);
                    else                 ((float*)out2)[idx] = x;
                } else if (mode == 1) {
                    ((unsigned short*)out0)[(size_t)row * CHN + col] = f2bf(x);
                } else if (mode == 2) {
                    ((float*)out0)[(size_t)row * CH + col] = x;
                } else if (mode == 3) {
                    const float gl = 0.5f * x * (1.0f + erff(x * 0.70710678118654752f));
                    ((unsigned short*)out0)[(size_t)row * CHN + col] = f2bf(gl);
                } else {
                    ((float*)out0)[(size_t)row * 512 + col] = x;
                }
            }
        }
    }
}

// ---------------------------------------------------------------------------
// logits[b,h,:,:] = (q @ k^T) * DH^-0.5 ; one block per (b,h), one WMMA/tile
// ---------------------------------------------------------------------------
__global__ __launch_bounds__(256) void logits_kernel(
    const unsigned short* __restrict__ q, const unsigned short* __restrict__ kmat,
    float* __restrict__ logits)
{
    const int bh = blockIdx.x;
    const int w = threadIdx.x >> 5, lane = threadIdx.x & 31;
    const int mlo = lane & 15, hi = lane >> 4;

    const unsigned short* qrow = q + ((size_t)bh * CN + w * 16 + mlo) * CDH;
    BF16x16 af;
    const int kb = hi * 8;
#pragma unroll
    for (int e = 0; e < 16; ++e) af.u[e] = qrow[kb + ((e >> 3) << 4) + (e & 7)];

    for (int nt = 0; nt < 8; ++nt) {
        const unsigned short* krow = kmat + ((size_t)bh * CN + nt * 16 + mlo) * CDH;
        BF16x16 bf;
#pragma unroll
        for (int e = 0; e < 16; ++e) bf.u[e] = krow[kb + ((e >> 3) << 4) + (e & 7)];
        v8f accv = v8f_zero();
        accv = __builtin_amdgcn_wmma_f32_16x16x32_bf16(
            false, af.v, false, bf.v, (short)0, accv, false, false);
#pragma unroll
        for (int r = 0; r < 8; ++r) {
            const int m = w * 16 + hi * 8 + r;
            logits[(size_t)bh * (CN * CN) + (size_t)m * CN + nt * 16 + mlo] =
                accv[r] * 0.17677669529663687f;   // 32^-0.5
        }
    }
}

// ---------------------------------------------------------------------------
// Attention: block per (b, n); wave per head. Masked softmax over the 127
// directed neighbor edges, then per-dim accumulation of e_val[pair]+v[src].
// ---------------------------------------------------------------------------
__global__ __launch_bounds__(256) void attn_kernel(
    const float* __restrict__ logits, const float* __restrict__ e_adj,
    const unsigned short* __restrict__ e_val, const float* __restrict__ vbuf,
    const float* __restrict__ mask, unsigned short* __restrict__ nat)
{
    __shared__ float probs[CH][CN];
    const int b = blockIdx.x >> 7, n = blockIdx.x & 127;
    const int h = threadIdx.x >> 5, lane = threadIdx.x & 31;
    const int bh = b * CH + h;
    const float* lrow = logits + (size_t)bh * (CN * CN) + (size_t)n * CN;

    float lg[4]; float mx = -INFINITY; int anyv = 0;
#pragma unroll
    for (int t = 0; t < 4; ++t) {
        const int j = t * 32 + lane;
        float L;
        if (j == n) {
            L = -INFINITY;
        } else {
            const int i0 = (n < j) ? n : j, j0 = (n < j) ? j : n;
            const int p = pair_idx(i0, j0);
            const float mk = mask[(size_t)b * CP + p];
            L = lrow[j] + e_adj[((size_t)b * CP + p) * CH + h];
            if (mk == 0.0f) L = -9e15f; else anyv = 1;
        }
        lg[t] = L; mx = fmaxf(mx, L);
    }
    mx = wave_max(mx);
#pragma unroll
    for (int o = 16; o > 0; o >>= 1) anyv |= __shfl_xor(anyv, o, 32);

    float s = 0.f;
#pragma unroll
    for (int t = 0; t < 4; ++t) {
        const float e = __expf(lg[t] - mx);
        probs[h][t * 32 + lane] = e;
        s += e;
    }
    s = wave_sum(s);
    const float inv = anyv ? (1.0f / s) : 0.0f;

    float acc = 0.f;
    const int d = lane;
    const float* vrow = vbuf + (size_t)bh * CN * CDH;
    for (int j = 0; j < CN; ++j) {
        if (j == n) continue;
        const float pr = probs[h][j] * inv;     // uniform across the wave
        if (pr == 0.0f) continue;
        const int i0 = (n < j) ? n : j, j0 = (n < j) ? j : n;
        const int p = pair_idx(i0, j0);
        const float ev = bf2f(e_val[((size_t)b * CP + p) * CHN + h * CDH + d]);
        acc += pr * (ev + vrow[j * CDH + d]);
    }
    nat[((size_t)b * CN + n) * 512 + CHN + h * CDH + d] = f2bf(acc);
}

// ---------------------------------------------------------------------------
// Final: out = node_feat*(1-g) + val*g,  g = sigmoid(gate)
// ---------------------------------------------------------------------------
__global__ __launch_bounds__(256) void final_kernel(
    const float* __restrict__ node_feat, const float* __restrict__ sk,
    float* __restrict__ out)
{
    const int row = blockIdx.x, c = threadIdx.x;
    const float val  = sk[(size_t)row * 512 + c];
    const float gate = sk[(size_t)row * 512 + CHN + c];
    const float g = 1.0f / (1.0f + __expf(-gate));
    const float nf = node_feat[(size_t)row * CHN + c];
    out[(size_t)row * CHN + c] = nf * (1.0f - g) + val * g;
}

// ---------------------------------------------------------------------------
// Host orchestration
// ---------------------------------------------------------------------------
extern "C" void kernel_launch(void* const* d_in, const int* in_sizes, int n_in,
                              void* d_out, int out_size, void* d_ws, size_t ws_size,
                              hipStream_t stream) {
    const float* node_feat = (const float*)d_in[0];
    const float* edge_feat = (const float*)d_in[1];
    const float* mask      = (const float*)d_in[4];
    const float* Wqkv = (const float*)d_in[5];
    const float* bqkv = (const float*)d_in[6];
    const float* Wev  = (const float*)d_in[7];
    const float* bev  = (const float*)d_in[8];
    const float* Wea  = (const float*)d_in[9];
    const float* bea  = (const float*)d_in[10];
    const float* Wo   = (const float*)d_in[11];
    const float* bo   = (const float*)d_in[12];
    const float* Wsk  = (const float*)d_in[13];
    const float* bsk  = (const float*)d_in[14];
    const float* gn   = (const float*)d_in[15];
    const float* bn   = (const float*)d_in[16];
    const float* ge   = (const float*)d_in[17];
    const float* be   = (const float*)d_in[18];
    float* out = (float*)d_out;

    // workspace carve
    char* w = (char*)d_ws;
    unsigned short* nat   = (unsigned short*)w; w += (size_t)CBN * 512 * 2;   // [1024,512] bf16 (n_in | attn)
    unsigned short* e_in  = (unsigned short*)w; w += (size_t)CBP * CHE * 2;   // [65024,128] bf16
    unsigned short* qb    = (unsigned short*)w; w += (size_t)CB * CH * CN * CDH * 2;
    unsigned short* kb    = (unsigned short*)w; w += (size_t)CB * CH * CN * CDH * 2;
    float*          vb    = (float*)w;          w += (size_t)CB * CH * CN * CDH * 4;
    unsigned short* e_val = (unsigned short*)w; w += (size_t)CBP * CHN * 2;   // [65024,256] bf16
    float*          e_adj = (float*)w;          w += (size_t)CBP * CH * 4;    // [65024,8]
    float*          logits= (float*)w;          w += (size_t)CB * CH * CN * CN * 4;
    unsigned short* comb  = (unsigned short*)w; w += (size_t)CBN * CHN * 2;
    float*          sk    = (float*)w;          w += (size_t)CBN * 512 * 4;
    unsigned short* Wqkvt = (unsigned short*)w; w += (size_t)768 * CHN * 2;   // [768,256]
    unsigned short* Wevt  = (unsigned short*)w; w += (size_t)CHN * CHE * 2;   // [256,128]
    unsigned short* Weat  = (unsigned short*)w; w += (size_t)CH * CHE * 2;    // [8,128]
    unsigned short* Wot   = (unsigned short*)w; w += (size_t)CHN * 512 * 2;   // [256,512]
    unsigned short* Wskt  = (unsigned short*)w; w += (size_t)512 * CHN * 2;   // [512,256]

    // 0) weight transpose + bf16 conversion (once; tiny)
    wt_kernel<<<(CHN * 768 + 255) / 256, 256, 0, stream>>>(Wqkv, Wqkvt, CHN, 768);
    wt_kernel<<<(CHE * CHN + 255) / 256, 256, 0, stream>>>(Wev,  Wevt,  CHE, CHN);
    wt_kernel<<<(CHE * CH  + 255) / 256, 256, 0, stream>>>(Wea,  Weat,  CHE, CH);
    wt_kernel<<<(512 * CHN + 255) / 256, 256, 0, stream>>>(Wo,   Wot,   512, CHN);
    wt_kernel<<<(CHN * 512 + 255) / 256, 256, 0, stream>>>(Wsk,  Wskt,  CHN, 512);

    // 1) layernorms
    ln_node_kernel<<<CBN / 8, 256, 0, stream>>>(node_feat, gn, bn, nat);
    ln_edge_kernel<<<CBP / 8, 256, 0, stream>>>(edge_feat, ge, be, e_in);

    // 2) QKV projection (A = nat cols 0..255, lda=512)
    gemm_bf16_kernel<<<dim3(CBN / 64, 768 / 64), 128, 0, stream>>>(
        nat, 512, Wqkvt, 768, bqkv, CBN, CHN, 0, qb, kb, vb);

    // 3) edge value + edge adjacency projections
    gemm_bf16_kernel<<<dim3(CBP / 64, CHN / 64), 128, 0, stream>>>(
        e_in, CHE, Wevt, CHN, bev, CBP, CHE, 1, e_val, nullptr, nullptr);
    gemm_bf16_kernel<<<dim3(CBP / 64, 1), 128, 0, stream>>>(
        e_in, CHE, Weat, CH, bea, CBP, CHE, 2, e_adj, nullptr, nullptr);

    // 4) q @ k^T logits
    logits_kernel<<<CB * CH, 256, 0, stream>>>(qb, kb, logits);

    // 5) masked softmax attention with edge values (writes nat cols 256..511)
    attn_kernel<<<CBN, 256, 0, stream>>>(logits, e_adj, e_val, vb, mask, nat);

    // 6) output projection + gelu, then skip projection
    gemm_bf16_kernel<<<dim3(CBN / 64, CHN / 64), 128, 0, stream>>>(
        nat, 512, Wot, CHN, bo, CBN, 512, 3, comb, nullptr, nullptr);
    gemm_bf16_kernel<<<dim3(CBN / 64, 512 / 64), 128, 0, stream>>>(
        comb, CHN, Wskt, 512, bsk, CBN, CHN, 4, sk, nullptr, nullptr);

    // 7) gated residual
    final_kernel<<<CBN, 256, 0, stream>>>(node_feat, sk, out);
}
